// TimeSeriesModel_85727547228528
// MI455X (gfx1250) — matrix operand order repaired and checked by reading, not compiled, not run on hardware
//
#include <hip/hip_runtime.h>
#include <hip/hip_bf16.h>
#include <math.h>

// MI455X / gfx1250 — wave32, WMMA 16x16x32 f16->f32.
typedef __attribute__((ext_vector_type(16))) _Float16 v16h;
typedef __attribute__((ext_vector_type(8)))  _Float16 v8h;
typedef __attribute__((ext_vector_type(8)))  float    v8f;

#define HID 64
#define G3 192            // 3*HID (r,z,n gates)
#define BATCH 16
#define TT 1024
#define NWAVE 12          // one wave per 16-column tile of the 192 gate columns
#define NTHREADS (NWAVE*32)

#define LOG2E  1.442695041f

// Hardware transcendental paths (v_exp_f32 + v_rcp_f32), no IEEE-divide chain.
__device__ __forceinline__ float fast_sigmoid(float x) {
    float e = __builtin_amdgcn_exp2f(x * -LOG2E);            // 2^(-x*log2e) = e^-x
    return __builtin_amdgcn_rcpf(1.0f + e);
}
__device__ __forceinline__ float fast_tanh(float x) {
    float e = __builtin_amdgcn_exp2f(x * (2.0f * LOG2E));    // e^(2x)
    return 1.0f - 2.0f * __builtin_amdgcn_rcpf(e + 1.0f);
}

__device__ __forceinline__ v8f wmma_f16(v16h a, v16h b, v8f c) {
    // (neg_a, A, neg_b, B, c_mod, C, reuse_a, reuse_b)
    return __builtin_amdgcn_wmma_f32_16x16x32_f16(false, a, false, b, (short)0, c, false, false);
}

// A fragment (16x32 f16, MxK) from row-major LDS matrix [16][64].
// lane&15 = M row ; (lane>>4)*8 = K sub-chunk per ISA layout:
//   halves 0..7  = K = kb*32 + koff + 0..7
//   halves 8..15 = K = kb*32 + koff + 16..23
__device__ __forceinline__ v16h load_a_frag(const _Float16* rowbase, int kb, int koff) {
    const v8h lo = *(const v8h*)(rowbase + kb * 32 + koff);
    const v8h hi = *(const v8h*)(rowbase + kb * 32 + koff + 16);
    v16h a;
#pragma unroll
    for (int i = 0; i < 8; ++i) { a[i] = lo[i]; a[i + 8] = hi[i]; }
    return a;
}

// B fragment (32x16 f16, KxN) for gi = x @ W^T  =>  B[k][n] = W[n][k].
// lane&15 = N column n ; halves i<8 = W[n][kb*32 + (lane>>4)*8 + i], i>=8 = W[n][kb*32 + (lane>>4)*8 + 8 + i] (matches A).
__device__ __forceinline__ v16h load_b_frag(const float* W, int l, int n, int kb, int kh) {
    const float* p = W + ((size_t)(l * G3 + n)) * HID + kb * 32 + kh / 2;
    v16h b;
#pragma unroll
    for (int i = 0; i < 16; ++i) b[i] = (_Float16)p[i < 8 ? i : i + 8];
    return b;
}

// ---------------------------------------------------------------------------
// Context GRU: one persistent workgroup, batch-16 GEMMs via WMMA,
// weights VGPR-resident across the whole 2048-step scan.
// ---------------------------------------------------------------------------
__global__ void __launch_bounds__(NTHREADS)
ctx_gru_kernel(const float* __restrict__ text, const float* __restrict__ audio,
               const float* __restrict__ Wih, const float* __restrict__ Whh,
               const float* __restrict__ bih, const float* __restrict__ bhh,
               float* __restrict__ out)
{
    __shared__ __align__(16) float    sh_gi[BATCH * G3];
    __shared__ __align__(16) float    sh_gh[BATCH * G3];
    __shared__ __align__(16) _Float16 sh_x[BATCH * HID];
    __shared__ __align__(16) float    sh_h[3][BATCH * HID];
    __shared__ __align__(16) _Float16 sh_hh[3][BATCH * HID];
    __shared__ __align__(16) int      sh_wflag[NWAVE];

    const int tid  = threadIdx.x;
    const int wave = tid >> 5;
    const int lane = tid & 31;
    const int nloc = lane & 15;
    const int gcol = wave * 16 + nloc;      // gate column 0..191
    const int koff = (lane >> 4) * 8;       // A K-chunk
    const int kh   = (lane >> 4) * 16;      // B K-chunk
    const int mrow = (lane >> 4) * 8;       // C M base

    for (int i = tid; i < 3 * BATCH * HID; i += NTHREADS) {
        int l = i / (BATCH * HID), r = i % (BATCH * HID);
        sh_h[l][r] = 0.f; sh_hh[l][r] = (_Float16)0.f;
    }

    // VGPR-resident weight fragments + biases folded into accumulator init.
    v16h wi[3][2], wh[3][2];
    float bi[3], bh[3];
#pragma unroll
    for (int l = 0; l < 3; ++l) {
#pragma unroll
        for (int kb = 0; kb < 2; ++kb) {
            wi[l][kb] = load_b_frag(Wih, l, gcol, kb, kh);
            wh[l][kb] = load_b_frag(Whh, l, gcol, kb, kh);
        }
        bi[l] = bih[l * G3 + gcol];
        bh[l] = bhh[l * G3 + gcol];
    }
    __syncthreads();

    const int OUTAUD = BATCH * TT * HID;

    for (int s = 0; s < 2 * TT; ++s) {
        // Load step input (f32 -> f16) + per-wave any(x != 0) reduction.
        int nz = 0;
        for (int idx = tid; idx < BATCH * HID; idx += NTHREADS) {
            int m = idx >> 6, c = idx & 63;
            float v = (s < TT) ? text[(m * TT + s) * HID + c]
                               : audio[(m * TT + (s - TT)) * HID + c];
            sh_x[idx] = (_Float16)v;
            nz |= (v != 0.f);
        }
        int wany = __any(nz);
        if (lane == 0) sh_wflag[wave] = wany;   // overwritten each step, no reset
        __syncthreads();
        int act = 0;
#pragma unroll
        for (int w = 0; w < NWAVE; ++w) act |= sh_wflag[w];
        const bool active = (act != 0);          // block-uniform -> EXEC full for WMMA

        if (active) {
#pragma unroll
            for (int l = 0; l < 3; ++l) {
                v8f ai, ah;
#pragma unroll
                for (int i = 0; i < 8; ++i) { ai[i] = bi[l]; ah[i] = bh[l]; }
                const _Float16* xr = sh_x + nloc * HID;
                const _Float16* hr = sh_hh[l] + nloc * HID;
                ai = wmma_f16(load_a_frag(xr, 0, koff), wi[l][0], ai);
                ai = wmma_f16(load_a_frag(xr, 1, koff), wi[l][1], ai);
                ah = wmma_f16(load_a_frag(hr, 0, koff), wh[l][0], ah);
                ah = wmma_f16(load_a_frag(hr, 1, koff), wh[l][1], ah);
#pragma unroll
                for (int r = 0; r < 8; ++r) {
                    sh_gi[(mrow + r) * G3 + gcol] = ai[r];
                    sh_gh[(mrow + r) * G3 + gcol] = ah[r];
                }
                __syncthreads();

                for (int idx = tid; idx < BATCH * HID; idx += NTHREADS) {
                    int m = idx >> 6, c = idx & 63;
                    float ir = sh_gi[m * G3 + c];
                    float iz = sh_gi[m * G3 + HID + c];
                    float in = sh_gi[m * G3 + 2 * HID + c];
                    float hrv = sh_gh[m * G3 + c];
                    float hz = sh_gh[m * G3 + HID + c];
                    float hn = sh_gh[m * G3 + 2 * HID + c];
                    float rg = fast_sigmoid(ir + hrv);
                    float zg = fast_sigmoid(iz + hz);
                    float ng = fast_tanh(in + rg * hn);
                    float hp = sh_h[l][idx];
                    float nh = (1.f - zg) * ng + zg * hp;
                    sh_h[l][idx] = nh;
                    sh_hh[l][idx] = (_Float16)nh;
                    if (l < 2) {
                        sh_x[idx] = (_Float16)nh;     // next layer input
                    } else {
                        int o = (s < TT) ? (m * TT + s) * HID + c
                                         : OUTAUD + (m * TT + (s - TT)) * HID + c;
                        out[o] = 0.5f * nh;           // 0.5 * context part
                    }
                }
                __syncthreads();
            }
        } else {
            for (int idx = tid; idx < BATCH * HID; idx += NTHREADS) {
                int m = idx >> 6, c = idx & 63;
                int o = (s < TT) ? (m * TT + s) * HID + c
                                 : OUTAUD + (m * TT + (s - TT)) * HID + c;
                out[o] = 0.f;
            }
            __syncthreads();
        }
    }
}

// ---------------------------------------------------------------------------
// Sample GRU: 32768-step serial scan. Male/female chains packed as
// M-rows 0/1 of the same WMMA tiles (rows 2..15 stay zero).
// ---------------------------------------------------------------------------
__global__ void __launch_bounds__(NTHREADS)
smp_gru_kernel(const float* __restrict__ text, const float* __restrict__ audio,
               const int* __restrict__ gender,
               const float* __restrict__ Wih, const float* __restrict__ Whh,
               const float* __restrict__ bih, const float* __restrict__ bhh,
               float* __restrict__ out)
{
    __shared__ __align__(16) float    sh_gi[2 * G3];
    __shared__ __align__(16) float    sh_gh[2 * G3];
    __shared__ __align__(16) _Float16 sh_x[BATCH * HID];     // rows 0,1 live; rest zero
    __shared__ __align__(16) _Float16 sh_hb[3][BATCH * HID]; // h banks (f16 A source)
    __shared__ __align__(16) float    sh_hf[3][2 * HID];     // h banks (f32)
    __shared__ __align__(16) int      sh_wflag[NWAVE];

    const int tid  = threadIdx.x;
    const int wave = tid >> 5;
    const int lane = tid & 31;
    const int nloc = lane & 15;
    const int gcol = wave * 16 + nloc;
    const int koff = (lane >> 4) * 8;
    const int kh   = (lane >> 4) * 16;

    for (int i = tid; i < BATCH * HID; i += NTHREADS) sh_x[i] = (_Float16)0.f;
    for (int i = tid; i < 3 * BATCH * HID; i += NTHREADS)
        sh_hb[i / (BATCH * HID)][i % (BATCH * HID)] = (_Float16)0.f;
    for (int i = tid; i < 3 * 2 * HID; i += NTHREADS)
        sh_hf[i / (2 * HID)][i % (2 * HID)] = 0.f;

    v16h wi[3][2], wh[3][2];
    float bi[3], bh[3];
#pragma unroll
    for (int l = 0; l < 3; ++l) {
#pragma unroll
        for (int kb = 0; kb < 2; ++kb) {
            wi[l][kb] = load_b_frag(Wih, l, gcol, kb, kh);
            wh[l][kb] = load_b_frag(Whh, l, gcol, kb, kh);
        }
        bi[l] = bih[l * G3 + gcol];
        bh[l] = bhh[l * G3 + gcol];
    }
    __syncthreads();

    const int OUTAUD = BATCH * TT * HID;

    for (int step = 0; step < BATCH * 2 * TT; ++step) {
        const int b    = step >> 11;      // / (2*TT)
        const int rr   = step & 2047;
        const int strm = rr >> 10;        // 0=text, 1=audio
        const int t    = rr & 1023;
        const float* xp = strm ? (audio + (size_t)(b * TT + t) * HID)
                               : (text  + (size_t)(b * TT + t) * HID);

        int nz = 0;
        if (tid < HID) {
            float v = xp[tid];
            _Float16 hv = (_Float16)v;
            sh_x[tid] = hv;          // row 0 (male chain input)
            sh_x[HID + tid] = hv;    // row 1 (female chain input)
            nz = (v != 0.f);
        }
        int wany = __any(nz);
        if (lane == 0) sh_wflag[wave] = wany;
        __syncthreads();
        int act = 0;
#pragma unroll
        for (int w = 0; w < NWAVE; ++w) act |= sh_wflag[w];
        const bool active = (act != 0);
        const bool male = (gender[b * TT + t] == 0);

        if (active) {
#pragma unroll
            for (int l = 0; l < 3; ++l) {
                v8f ai, ah;
#pragma unroll
                for (int i = 0; i < 8; ++i) { ai[i] = bi[l]; ah[i] = bh[l]; }
                const _Float16* xr = sh_x + nloc * HID;
                const _Float16* hr = sh_hb[l] + nloc * HID;
                ai = wmma_f16(load_a_frag(xr, 0, koff), wi[l][0], ai);
                ai = wmma_f16(load_a_frag(xr, 1, koff), wi[l][1], ai);
                ah = wmma_f16(load_a_frag(hr, 0, koff), wh[l][0], ah);
                ah = wmma_f16(load_a_frag(hr, 1, koff), wh[l][1], ah);
                if (lane < 16) {                     // only M=0,1 matter
                    sh_gi[0 * G3 + gcol] = ai[0];
                    sh_gi[1 * G3 + gcol] = ai[1];
                    sh_gh[0 * G3 + gcol] = ah[0];
                    sh_gh[1 * G3 + gcol] = ah[1];
                }
                __syncthreads();

                if (tid < 2 * HID) {
                    int m = tid >> 6, c = tid & 63;  // m: 0=male chain, 1=female chain
                    float ir = sh_gi[m * G3 + c];
                    float iz = sh_gi[m * G3 + HID + c];
                    float in = sh_gi[m * G3 + 2 * HID + c];
                    float hrv = sh_gh[m * G3 + c];
                    float hz = sh_gh[m * G3 + HID + c];
                    float hn = sh_gh[m * G3 + 2 * HID + c];
                    float rg = fast_sigmoid(ir + hrv);
                    float zg = fast_sigmoid(iz + hz);
                    float ng = fast_tanh(in + rg * hn);
                    float hp = sh_hf[l][m * HID + c];
                    float nh = (1.f - zg) * ng + zg * hp;   // candidate
                    bool upd = (m == 0) ? male : !male;     // active already true
                    float bank = upd ? nh : hp;
                    sh_hf[l][m * HID + c] = bank;
                    sh_hb[l][m * HID + c] = (_Float16)bank;
                    sh_x[m * HID + c] = (_Float16)nh;       // next layer uses candidate
                    if (l == 2 && m == 1) {                 // output = new female bank
                        size_t o = strm ? (size_t)OUTAUD + (size_t)(b * TT + t) * HID + c
                                        : (size_t)(b * TT + t) * HID + c;
                        out[o] += 0.5f * bank;
                    }
                }
                __syncthreads();
            }
        }
    }
}

extern "C" void kernel_launch(void* const* d_in, const int* in_sizes, int n_in,
                              void* d_out, int out_size, void* d_ws, size_t ws_size,
                              hipStream_t stream) {
    (void)in_sizes; (void)n_in; (void)d_ws; (void)ws_size; (void)out_size;
    const float* text  = (const float*)d_in[0];
    const float* audio = (const float*)d_in[1];
    const int*   gend  = (const int*)d_in[2];
    const float* cWih  = (const float*)d_in[3];
    const float* cWhh  = (const float*)d_in[4];
    const float* cbih  = (const float*)d_in[5];
    const float* cbhh  = (const float*)d_in[6];
    const float* sWih  = (const float*)d_in[7];
    const float* sWhh  = (const float*)d_in[8];
    const float* sbih  = (const float*)d_in[9];
    const float* sbhh  = (const float*)d_in[10];
    float* out = (float*)d_out;

    // Stream-ordered: ctx writes 0.5*context to all of d_out, then sample
    // accumulates 0.5*sample. Both are persistent single-WGP serial scans.
    ctx_gru_kernel<<<1, NTHREADS, 0, stream>>>(text, audio, cWih, cWhh, cbih, cbhh, out);
    smp_gru_kernel<<<1, NTHREADS, 0, stream>>>(text, audio, gend, sWih, sWhh, sbih, sbhh, out);
}
